// SelfAttention2d_80298708566372
// MI455X (gfx1250) — matrix operand order repaired and checked by reading, not compiled, not run on hardware
//
#include <hip/hip_runtime.h>
#include <hip/hip_bf16.h>

#define CH 256
#define NTOK 4096
#define BATCH 8
#define GROUPS 8

typedef __attribute__((ext_vector_type(16))) _Float16 v16h;
typedef __attribute__((ext_vector_type(8)))  _Float16 v8h;
typedef __attribute__((ext_vector_type(8)))  float    v8f;

// ---- fragment helpers -------------------------------------------------------
// A-operand (16x32 f16): lane<16 holds K 0..15, lane>=16 holds K 16..31 (contiguous)
// -> caller passes p = row_base + chunk*32 + hi*16 ; we read p[0..15]
static __device__ __forceinline__ v16h ld_fragA(const _Float16* p) {
  return *(const v16h*)(p);
}
// B-operand (32x16 f16): lane<16 holds K 0..15, lane>=16 holds K 16..31 (contiguous)
static __device__ __forceinline__ v16h ld_fragB(const _Float16* p) {
  return *(const v16h*)(p);
}
static __device__ __forceinline__ v8f wmma_f16(v16h a, v16h b, v8f c) {
  return __builtin_amdgcn_wmma_f32_16x16x32_f16(false, a, false, b, (short)0, c,
                                                false, false);
}
static __device__ __forceinline__ v8f vzero8f() {
  v8f z;
#pragma unroll
  for (int i = 0; i < 8; ++i) z[i] = 0.f;
  return z;
}

// ---- phase 0: fp32 -> fp16 weight conversion --------------------------------
__global__ void k_f32_to_f16(const float* __restrict__ src,
                             _Float16* __restrict__ dst, int n) {
  int i = blockIdx.x * blockDim.x + threadIdx.x;
  if (i < n) dst[i] = (_Float16)src[i];
}

// ---- phase 1: GroupNorm(8, 256) -> yT[b][n][c] f16 (token-major) ------------
__global__ void __launch_bounds__(256)
k_groupnorm(const float* __restrict__ x, const float* __restrict__ gw,
            const float* __restrict__ gb, _Float16* __restrict__ yT) {
  const int cpg = CH / GROUPS;  // 32
  int b = blockIdx.x / GROUPS, g = blockIdx.x % GROUPS;
  const float* xp = x + ((size_t)b * CH + (size_t)g * cpg) * NTOK;
  float s = 0.f, ss = 0.f;
  const float4* xp4 = (const float4*)xp;
  for (int i = threadIdx.x; i < (cpg * NTOK) / 4; i += blockDim.x) {
    float4 v = xp4[i];
    s += v.x + v.y + v.z + v.w;
    ss += v.x * v.x + v.y * v.y + v.z * v.z + v.w * v.w;
  }
  __shared__ float sh0[256], sh1[256];
  sh0[threadIdx.x] = s; sh1[threadIdx.x] = ss;
  __syncthreads();
  for (int off = 128; off > 0; off >>= 1) {
    if ((int)threadIdx.x < off) {
      sh0[threadIdx.x] += sh0[threadIdx.x + off];
      sh1[threadIdx.x] += sh1[threadIdx.x + off];
    }
    __syncthreads();
  }
  float inv = 1.f / (float)(cpg * NTOK);
  float mu = sh0[0] * inv;
  float var = sh1[0] * inv - mu * mu;
  float rs = rsqrtf(var + 1e-5f);
  _Float16* yb = yT + (size_t)b * NTOK * CH;
  for (int cc = 0; cc < cpg; ++cc) {
    int ch = g * cpg + cc;
    float wv = gw[ch] * rs;
    float bv = gb[ch] - mu * wv;
    const float* row = xp + (size_t)cc * NTOK;
    for (int t = threadIdx.x; t < NTOK; t += blockDim.x)
      yb[(size_t)t * CH + ch] = (_Float16)(row[t] * wv + bv);
  }
}

// ---- phase 2a: token-major GEMM (Q, K): out[b][t][o] = W[o][:]·y[:,t] + bias -
// wave job = one 16-token tile x 256 out-channels. D: lane = t, vgpr = o.
// Inner loop pipelined: next W fragment's loads in flight under current WMMA.
__global__ void __launch_bounds__(128)
k_gemm_tok(const _Float16* __restrict__ yT, const _Float16* __restrict__ Wh,
           const float* __restrict__ bias, _Float16* __restrict__ out) {
  int wave = blockIdx.x * (blockDim.x >> 5) + ((int)threadIdx.x >> 5);
  int lane = threadIdx.x & 31;
  int lanelo = lane & 15, hi = lane >> 4;
  int b = wave >> 8;
  int t0 = (wave & 255) * 16;
  const _Float16* yb = yT + ((size_t)b * NTOK + t0) * CH;
  v8f acc[16];
#pragma unroll
  for (int i = 0; i < 16; ++i) acc[i] = vzero8f();
#pragma unroll
  for (int ck = 0; ck < 8; ++ck) {
    const _Float16* wrow = Wh + (size_t)lanelo * CH + ck * 32 + hi * 16;
    v16h bf = ld_fragB(yb + (size_t)lanelo * CH + ck * 32 + hi * 16);
    v16h a0 = ld_fragA(wrow);
    v16h a1 = ld_fragA(wrow + (size_t)16 * CH);
#pragma unroll
    for (int ot = 0; ot < 16; ++ot) {
      v16h an = a0;
      if (ot < 14) an = ld_fragA(wrow + (size_t)((ot + 2) * 16) * CH);
      acc[ot] = wmma_f16(a0, bf, acc[ot]);
      a0 = a1; a1 = an;
    }
  }
  _Float16* ob = out + ((size_t)b * NTOK + t0 + lanelo) * CH;
#pragma unroll
  for (int ot = 0; ot < 16; ++ot) {
    v8f bb = *(const v8f*)(bias + ot * 16 + hi * 8);
    v8h h;
#pragma unroll
    for (int r = 0; r < 8; ++r) h[r] = (_Float16)(acc[ot][r] + bb[r]);
    *(v8h*)(ob + ot * 16 + hi * 8) = h;
  }
}

// ---- phase 2b: channel-major GEMM (V): outT[b][o][t] ------------------------
// wave job = 16 out-channels x 256 tokens. D: lane = o, vgpr = t. Pipelined.
__global__ void __launch_bounds__(128)
k_gemm_chan(const _Float16* __restrict__ yT, const _Float16* __restrict__ Wh,
            const float* __restrict__ bias, _Float16* __restrict__ outT) {
  int wave = blockIdx.x * (blockDim.x >> 5) + ((int)threadIdx.x >> 5);
  int lane = threadIdx.x & 31;
  int lanelo = lane & 15, hi = lane >> 4;
  int b = wave >> 8;
  int rem = wave & 255;
  int o0 = (rem >> 4) * 16;
  int t0 = (rem & 15) * 256;
  v8f acc[16];
#pragma unroll
  for (int i = 0; i < 16; ++i) acc[i] = vzero8f();
#pragma unroll
  for (int ck = 0; ck < 8; ++ck) {
    const _Float16* yrow =
        yT + ((size_t)b * NTOK + t0 + lanelo) * CH + ck * 32 + hi * 16;
    v16h bf = ld_fragB(Wh + (size_t)(o0 + lanelo) * CH + ck * 32 + hi * 16);
    v16h a0 = ld_fragA(yrow);
    v16h a1 = ld_fragA(yrow + (size_t)16 * CH);
#pragma unroll
    for (int tt = 0; tt < 16; ++tt) {
      v16h an = a0;
      if (tt < 14) an = ld_fragA(yrow + (size_t)((tt + 2) * 16) * CH);
      acc[tt] = wmma_f16(a0, bf, acc[tt]);
      a0 = a1; a1 = an;
    }
  }
  int o = o0 + lanelo;
  float bv = bias[o];
  _Float16* ob = outT + ((size_t)b * CH + o) * (size_t)NTOK;
#pragma unroll
  for (int tt = 0; tt < 16; ++tt) {
    v8h h;
#pragma unroll
    for (int r = 0; r < 8; ++r) h[r] = (_Float16)(acc[tt][r] + bv);
    *(v8h*)(ob + t0 + tt * 16 + hi * 8) = h;
  }
}

// ---- phase 3: flash attention. S^T = K·Q^T, online softmax, O^T += V^T·P^T --
// wave job = one 16-query tile; iterate 32-key blocks; aoT[b][t][c] f16 out.
__global__ void __launch_bounds__(128)
k_attn(const _Float16* __restrict__ Qh, const _Float16* __restrict__ Kh,
       const _Float16* __restrict__ VTh, _Float16* __restrict__ aoT) {
  int wave = blockIdx.x * (blockDim.x >> 5) + ((int)threadIdx.x >> 5);
  int lane = threadIdx.x & 31;
  int lanelo = lane & 15, hi = lane >> 4;
  int b = wave >> 8;
  int q0 = (wave & 255) * 16;

  // Preload Q fragments (B-operand layout), fold in 1/sqrt(256) = 1/16 scale.
  v16h qf[8];
  const _Float16* qb = Qh + ((size_t)b * NTOK + q0 + lanelo) * CH;
#pragma unroll
  for (int ck = 0; ck < 8; ++ck) {
    v16h t = ld_fragB(qb + ck * 32 + hi * 16);
#pragma unroll
    for (int i = 0; i < 16; ++i) t[i] = t[i] * (_Float16)0.0625f;
    qf[ck] = t;
  }

  v8f acc[16];  // O^T: 16 d-tiles of 16x16 (d = vgpr row, query m = lane col)
#pragma unroll
  for (int i = 0; i < 16; ++i) acc[i] = vzero8f();
  float mx = -1e30f, l = 0.f;

  const _Float16* Kb0 = Kh + (size_t)b * NTOK * CH;
  const _Float16* Vb0 = VTh + (size_t)b * CH * NTOK;

  for (int jb = 0; jb < NTOK; jb += 32) {
    // S^T tiles: st0 = keys jb..jb+15, st1 = keys jb+16..jb+31 (x 16 queries)
    v8f st0 = vzero8f(), st1 = vzero8f();
    const _Float16* kb = Kb0 + (size_t)(jb + lanelo) * CH + hi * 16;
    v16h k0 = ld_fragA(kb);
    v16h k1 = ld_fragA(kb + (size_t)16 * CH);
#pragma unroll
    for (int ck = 0; ck < 8; ++ck) {
      v16h n0 = k0, n1 = k1;
      if (ck < 7) {
        n0 = ld_fragA(kb + (ck + 1) * 32);
        n1 = ld_fragA(kb + (size_t)16 * CH + (ck + 1) * 32);
      }
      st0 = wmma_f16(k0, qf[ck], st0);
      st1 = wmma_f16(k1, qf[ck], st1);
      k0 = n0; k1 = n1;
    }
    // per-query block max: local over 16 j values, then partner lane (xor 16)
    float bm = -1e30f;
#pragma unroll
    for (int r = 0; r < 8; ++r) {
      bm = fmaxf(bm, st0[r]);
      bm = fmaxf(bm, st1[r]);
    }
    bm = fmaxf(bm, __shfl_xor(bm, 16, 32));
    float mnew = fmaxf(mx, bm);
    float corr = __expf(mx - mnew);
    float e0[8], e1[8];
    float rsum = 0.f;
#pragma unroll
    for (int r = 0; r < 8; ++r) {
      e0[r] = __expf(st0[r] - mnew);
      e1[r] = __expf(st1[r] - mnew);
      rsum += e0[r] + e1[r];
    }
    rsum += __shfl_xor(rsum, 16, 32);
    l = l * corr + rsum;
    mx = mnew;
    // Assemble P^T as B operand: lane<16 needs j 0..15, lane>=16 needs j 16..31
    v16h pf;
#pragma unroll
    for (int r = 0; r < 8; ++r) {
      float x0 = __shfl_xor(e0[r], 16, 32);  // partner's st0 exp (j = 8+r / r)
      float x1 = __shfl_xor(e1[r], 16, 32);  // partner's st1 exp
      float lo = hi ? x1 : e0[r];            // j = 16*hi + r
      float hv = hi ? e1[r] : x0;            // j = 16*hi + 8 + r
      pf[r] = (_Float16)lo;
      pf[r + 8] = (_Float16)hv;
    }
    // Issue first V fragment load, then hide it under the accumulator rescale.
    const _Float16* vb = Vb0 + (size_t)lanelo * NTOK + jb + hi * 16;
    v16h vf = ld_fragA(vb);
#pragma unroll
    for (int dt = 0; dt < 16; ++dt) {
#pragma unroll
      for (int r = 0; r < 8; ++r) acc[dt][r] *= corr;
    }
#pragma unroll
    for (int dt = 0; dt < 16; ++dt) {
      v16h vn = vf;
      if (dt < 15) vn = ld_fragA(vb + (size_t)((dt + 1) * 16) * NTOK);
      acc[dt] = wmma_f16(vf, pf, acc[dt]);
      vf = vn;
    }
  }
  float inv = 1.f / l;
  _Float16* ob = aoT + ((size_t)b * NTOK + q0 + lanelo) * CH;
#pragma unroll
  for (int dt = 0; dt < 16; ++dt) {
    v8h h;
#pragma unroll
    for (int r = 0; r < 8; ++r) h[r] = (_Float16)(acc[dt][r] * inv);
    *(v8h*)(ob + dt * 16 + hi * 8) = h;
  }
}

// ---- phase 4: out = x + wp·ao + bp, f32, [b][c][n] --------------------------
// wave job = one 16-token tile x 256 channels. D: lane = o, vgpr = t. Pipelined.
__global__ void __launch_bounds__(128)
k_proj(const _Float16* __restrict__ aoT, const _Float16* __restrict__ wph,
       const float* __restrict__ bp, const float* __restrict__ x,
       float* __restrict__ out) {
  int wave = blockIdx.x * (blockDim.x >> 5) + ((int)threadIdx.x >> 5);
  int lane = threadIdx.x & 31;
  int lanelo = lane & 15, hi = lane >> 4;
  int b = wave >> 8;
  int t0 = (wave & 255) * 16;
  v8f acc[16];
#pragma unroll
  for (int i = 0; i < 16; ++i) acc[i] = vzero8f();
  const _Float16* ab = aoT + ((size_t)b * NTOK + t0 + lanelo) * CH;
#pragma unroll
  for (int ck = 0; ck < 8; ++ck) {
    const _Float16* wrow = wph + (size_t)lanelo * CH + ck * 32 + hi * 16;
    v16h af = ld_fragA(ab + ck * 32 + hi * 16);
    v16h b0 = ld_fragB(wrow);
    v16h b1 = ld_fragB(wrow + (size_t)16 * CH);
#pragma unroll
    for (int ot = 0; ot < 16; ++ot) {
      v16h bn = b0;
      if (ot < 14) bn = ld_fragB(wrow + (size_t)((ot + 2) * 16) * CH);
      acc[ot] = wmma_f16(af, b0, acc[ot]);
      b0 = b1; b1 = bn;
    }
  }
#pragma unroll
  for (int ot = 0; ot < 16; ++ot) {
    int o = ot * 16 + lanelo;
    float bv = bp[o];
    size_t off = ((size_t)b * CH + o) * (size_t)NTOK + t0 + hi * 8;
    v8f xr = *(const v8f*)(x + off);
    v8f res;
#pragma unroll
    for (int r = 0; r < 8; ++r) res[r] = xr[r] + acc[ot][r] + bv;
    *(v8f*)(out + off) = res;
  }
}

// ---- host-side launcher -----------------------------------------------------
extern "C" void kernel_launch(void* const* d_in, const int* in_sizes, int n_in,
                              void* d_out, int out_size, void* d_ws,
                              size_t ws_size, hipStream_t stream) {
  (void)in_sizes; (void)n_in; (void)out_size; (void)ws_size;
  const float* x   = (const float*)d_in[0];
  const float* gnw = (const float*)d_in[1];
  const float* gnb = (const float*)d_in[2];
  const float* wq  = (const float*)d_in[3];
  const float* bq  = (const float*)d_in[4];
  const float* wk  = (const float*)d_in[5];
  const float* bk  = (const float*)d_in[6];
  const float* wv  = (const float*)d_in[7];
  const float* bv  = (const float*)d_in[8];
  const float* wp  = (const float*)d_in[9];
  const float* bp  = (const float*)d_in[10];
  float* out = (float*)d_out;

  const size_t nMat = (size_t)BATCH * NTOK * CH;  // 8388608 elements
  _Float16* ws  = (_Float16*)d_ws;
  _Float16* yT  = ws;              // [b][n][c] normalized, f16
  _Float16* Qh  = yT + nMat;       // [b][n][c]
  _Float16* Kh  = Qh + nMat;       // [b][n][c]
  _Float16* VTh = Kh + nMat;       // [b][c][n]
  _Float16* aoT = VTh + nMat;      // [b][n][c] attention output
  _Float16* wqh = aoT + nMat;      // 256x256 each
  _Float16* wkh = wqh + CH * CH;
  _Float16* wvh = wkh + CH * CH;
  _Float16* wph = wvh + CH * CH;

  k_f32_to_f16<<<(CH * CH) / 256, 256, 0, stream>>>(wq, wqh, CH * CH);
  k_f32_to_f16<<<(CH * CH) / 256, 256, 0, stream>>>(wk, wkh, CH * CH);
  k_f32_to_f16<<<(CH * CH) / 256, 256, 0, stream>>>(wv, wvh, CH * CH);
  k_f32_to_f16<<<(CH * CH) / 256, 256, 0, stream>>>(wp, wph, CH * CH);

  k_groupnorm<<<BATCH * GROUPS, 256, 0, stream>>>(x, gnw, gnb, yT);

  // 2048 wave-jobs each, 4 waves per 128-thread block -> 512 blocks
  k_gemm_tok<<<512, 128, 0, stream>>>(yT, wqh, bq, Qh);
  k_gemm_tok<<<512, 128, 0, stream>>>(yT, wkh, bk, Kh);
  k_gemm_chan<<<512, 128, 0, stream>>>(yT, wvh, bv, VTh);

  k_attn<<<512, 128, 0, stream>>>(Qh, Kh, VTh, aoT);

  k_proj<<<512, 128, 0, stream>>>(aoT, wph, bp, x, out);
}
